// MoE_Res_FCNN_48911087567476
// MI455X (gfx1250) — compile-verified, matrix operand order checked by reading, not run on hardware
//
#include <hip/hip_runtime.h>
#include <math.h>

// ---------------- problem constants ----------------
constexpr int N_TOK = 16384;
constexpr int C_IN  = 512;
constexpr int HIDN  = 1024;
constexpr int NE    = 8;
constexpr int KTOP  = 2;
constexpr int C_OUT = 512;
constexpr int NRES  = 3;

// ---------------- GEMM tile constants ----------------
constexpr int BM = 128, BN = 128, BK = 32;
constexpr int LDT = 40;          // padded LDS row (bf16 elems) -> conflict-free b128 frag loads

typedef __attribute__((ext_vector_type(16))) __bf16 v16bf;
typedef __attribute__((ext_vector_type(8)))  float  v8f;
typedef __attribute__((ext_vector_type(4)))  int    v4i;

union FragU { uint4 q[2]; v16bf v; };

// ---- CDNA5 async global->LDS path (probe-guarded with sync fallback) ----
#if defined(__has_builtin)
#if __has_builtin(__builtin_amdgcn_global_load_async_to_lds_b128) && \
    __has_builtin(__builtin_amdgcn_s_wait_asynccnt)
#define USE_ASYNC_LDS 1
#else
#define USE_ASYNC_LDS 0
#endif
#else
#define USE_ASYNC_LDS 0
#endif

__device__ __forceinline__ void async_ld_b128(const unsigned short* gsrc, unsigned short* ldst) {
#if USE_ASYNC_LDS
    __builtin_amdgcn_global_load_async_to_lds_b128(
        (__attribute__((address_space(1))) v4i*)(void*)gsrc,
        (__attribute__((address_space(3))) v4i*)ldst, 0, 0);
#else
    *(uint4*)ldst = *(const uint4*)gsrc;
#endif
}
__device__ __forceinline__ void wait_async() {
#if USE_ASYNC_LDS
    __builtin_amdgcn_s_wait_asynccnt(0);
#endif
}

__device__ __forceinline__ unsigned short f2bf(float f) {
    unsigned u = __float_as_uint(f);
    unsigned r = u + 0x7FFFu + ((u >> 16) & 1u);   // round-to-nearest-even
    return (unsigned short)(r >> 16);
}

// ---------------- elementwise conversions ----------------
__global__ void conv_f2b(const float* __restrict__ in, unsigned short* __restrict__ out, long total) {
    long i = (long)blockIdx.x * blockDim.x + threadIdx.x;
    if (i < total) out[i] = f2bf(in[i]);
}
__global__ void zero_f32(float* __restrict__ p, long total) {
    long i = (long)blockIdx.x * blockDim.x + threadIdx.x;
    if (i < total) p[i] = 0.0f;
}
// convert + transpose: in is [mats][K][Nn] fp32 row-major; out is [mats][Nn][K] bf16
__global__ void convT(const float* __restrict__ in, unsigned short* __restrict__ out,
                      int K, int Nn, long total) {
    long i = (long)blockIdx.x * blockDim.x + threadIdx.x;
    if (i >= total) return;
    long ms  = (long)K * Nn;
    int  mat = (int)(i / ms);
    long r   = i - (long)mat * ms;
    int  n   = (int)(r / K);
    int  k   = (int)(r - (long)n * K);
    out[i] = f2bf(in[(long)mat * ms + (long)k * Nn + n]);
}

// ---------------- gating: logits, top-2, softmax ----------------
__global__ __launch_bounds__(256) void gating_kernel(
        const float* __restrict__ x, const float* __restrict__ wg,
        int* __restrict__ top_idx, float* __restrict__ top_gate) {
    __shared__ float wgs[C_IN * NE];       // 16 KB
    __shared__ float red[256][NE];         // 8 KB
    int tid = threadIdx.x;
    for (int i = tid; i < C_IN * NE; i += 256) wgs[i] = wg[i];
    __syncthreads();
    int tl = tid >> 2, sub = tid & 3;       // 4 threads per token
    long n = (long)blockIdx.x * 64 + tl;
    float acc[NE];
    #pragma unroll
    for (int e = 0; e < NE; ++e) acc[e] = 0.f;
    const float* xr = x + n * C_IN;
    for (int c = 0; c < C_IN / 16; ++c) {
        int i0 = c * 16 + sub * 4;
        float4 xv = *(const float4*)(xr + i0);
        float xs[4] = {xv.x, xv.y, xv.z, xv.w};
        #pragma unroll
        for (int t = 0; t < 4; ++t) {
            const float* wr = &wgs[(i0 + t) * NE];
            #pragma unroll
            for (int e = 0; e < NE; ++e) acc[e] += xs[t] * wr[e];
        }
    }
    #pragma unroll
    for (int e = 0; e < NE; ++e) red[tid][e] = acc[e];
    __syncthreads();
    if (sub == 0) {
        float lg[NE];
        #pragma unroll
        for (int e = 0; e < NE; ++e)
            lg[e] = red[tid][e] + red[tid + 1][e] + red[tid + 2][e] + red[tid + 3][e];
        int i0 = 0; float l0 = lg[0];
        for (int e = 1; e < NE; ++e) if (lg[e] > l0) { l0 = lg[e]; i0 = e; }
        int i1 = (i0 == 0) ? 1 : 0; float l1 = lg[i1];
        for (int e = 0; e < NE; ++e) if (e != i0 && lg[e] > l1) { l1 = lg[e]; i1 = e; }
        float e1 = expf(l1 - l0);
        float dinv = 1.f / (1.f + e1);
        top_idx [n * 2]     = i0;  top_idx [n * 2 + 1] = i1;
        top_gate[n * 2]     = dinv; top_gate[n * 2 + 1] = e1 * dinv;
    }
}

// ---------------- aux loss (deterministic fixed-order reduction) ----------------
__device__ __forceinline__ float cv_sq(const float v[NE]) {
    float m = 0.f;
    for (int e = 0; e < NE; ++e) m += v[e];
    m *= (1.f / NE);
    float var = 0.f;
    for (int e = 0; e < NE; ++e) { float d = v[e] - m; var += d * d; }
    var *= (1.f / NE);
    return var / (m * m + 1e-10f);
}
__global__ __launch_bounds__(256) void moe_stats(
        const int* __restrict__ top_idx, const float* __restrict__ top_gate,
        float* __restrict__ aux_out) {
    __shared__ float s[256][NE];
    int tid = threadIdx.x;
    float imp[NE], cnt[NE];
    #pragma unroll
    for (int e = 0; e < NE; ++e) { imp[e] = 0.f; cnt[e] = 0.f; }
    for (int i = tid; i < N_TOK * KTOP; i += 256) {
        int e = top_idx[i]; float g = top_gate[i];
        imp[e] += g; cnt[e] += (g > 0.f) ? 1.f : 0.f;
    }
    #pragma unroll
    for (int e = 0; e < NE; ++e) s[tid][e] = imp[e];
    __syncthreads();
    for (int st = 128; st > 0; st >>= 1) {
        if (tid < st) { for (int e = 0; e < NE; ++e) s[tid][e] += s[tid + st][e]; }
        __syncthreads();
    }
    float impT[NE];
    if (tid == 0) { for (int e = 0; e < NE; ++e) impT[e] = s[0][e]; }
    __syncthreads();
    #pragma unroll
    for (int e = 0; e < NE; ++e) s[tid][e] = cnt[e];
    __syncthreads();
    for (int st = 128; st > 0; st >>= 1) {
        if (tid < st) { for (int e = 0; e < NE; ++e) s[tid][e] += s[tid + st][e]; }
        __syncthreads();
    }
    if (tid == 0) {
        float cntT[NE];
        for (int e = 0; e < NE; ++e) cntT[e] = s[0][e];
        *aux_out = 0.01f * (cv_sq(impT) + cv_sq(cntT));
    }
}

// ---------------- deterministic counting sort: tokens -> expert buckets ----------------
__global__ __launch_bounds__(256) void bucket_sort(
        const int* __restrict__ top_idx, const float* __restrict__ top_gate,
        int* __restrict__ row_token, float* __restrict__ row_gate,
        int* __restrict__ eoff, int* __restrict__ ecnt) {
    __shared__ int h[256][NE];
    __shared__ int totals[NE], ebase[NE];
    int tid = threadIdx.x;
    const int CH = N_TOK / 256;
    int local[NE];
    #pragma unroll
    for (int e = 0; e < NE; ++e) local[e] = 0;
    int t0 = tid * CH;
    for (int t = 0; t < CH; ++t) {
        int n = t0 + t;
        local[top_idx[n * 2]]++;
        local[top_idx[n * 2 + 1]]++;
    }
    #pragma unroll
    for (int e = 0; e < NE; ++e) h[tid][e] = local[e];
    __syncthreads();
    if (tid < NE) {                         // exclusive column scan (in place)
        int e = tid, run = 0;
        for (int t = 0; t < 256; ++t) { int tmp = h[t][e]; h[t][e] = run; run += tmp; }
        totals[e] = run;
    }
    __syncthreads();
    if (tid == 0) {
        int base = 0;
        for (int e = 0; e < NE; ++e) { ebase[e] = base; base += totals[e]; }
    }
    __syncthreads();
    if (tid < NE) { eoff[tid] = ebase[tid]; ecnt[tid] = totals[tid]; }
    int pos[NE];
    #pragma unroll
    for (int e = 0; e < NE; ++e) pos[e] = ebase[e] + h[tid][e];
    for (int t = 0; t < CH; ++t) {
        int n = t0 + t;
        for (int j = 0; j < KTOP; ++j) {
            int e = top_idx[n * 2 + j];
            int p = pos[e]++;
            row_token[p] = n;
            row_gate[p]  = top_gate[n * 2 + j];
        }
    }
}

// ---------------- bf16 WMMA GEMM, 128x128x32 tile, 8 waves, async double-buffer ----------------
// EPI: 0 = bf16(tanh(acc+bias))           -> outB
//      1 = atomicAdd(outF[token], gate*(acc+bias))   (expert combine, 2 commutative adds)
//      2 = t=tanh(acc+bias+resid) -> outF (f32) and outB (bf16)
//      3 = acc+bias -> outF
template<bool GATHER, int EPI>
__global__ __launch_bounds__(256) void gemm_wmma(
        const unsigned short* __restrict__ A, int lda,
        const unsigned short* __restrict__ B, long strideB,
        const float* __restrict__ bias, int strideBias,
        int K, int Nw,
        const int* __restrict__ grp_off, const int* __restrict__ grp_cnt,
        int tilesPerGroup, int M_dense,
        const int* __restrict__ row_token, const float* __restrict__ row_gate,
        float* __restrict__ outF, unsigned short* __restrict__ outB,
        const float* __restrict__ resid) {
    __shared__ unsigned short As[2][BM * LDT];   // 2 x 10 KB
    __shared__ unsigned short Bs[2][BN * LDT];   // 2 x 10 KB
    __shared__ int   rtok[BM];
    __shared__ float rgat[BM];

    const int tid  = threadIdx.x;
    const int g    = blockIdx.y / tilesPerGroup;
    const int tile = blockIdx.y % tilesPerGroup;
    int off, cnt;
    if (grp_off) { off = grp_off[g]; cnt = grp_cnt[g]; }
    else         { off = 0;          cnt = M_dense;    }
    const int row0 = off + tile * BM;
    if (row0 >= off + cnt) return;
    const int rows = min(BM, off + cnt - row0);
    const int nb0  = blockIdx.x * BN;

    const unsigned short* Bg = B + (long)g * strideB;
    const float* biasg = bias + (long)g * strideBias;

    // stage scatter metadata once (EPI==1)
    if (EPI == 1 && tid < rows) {
        rtok[tid] = row_token[row0 + tid];
        rgat[tid] = row_gate[row0 + tid];
    }

    // ---- hoisted per-thread staging addresses (gather index loaded once) ----
    const unsigned short* aSrc[2]; int aOff[2];
    const unsigned short* bSrc[2]; int bOff[2];
    #pragma unroll
    for (int it = 0; it < 2; ++it) {
        int idx = it * 256 + tid;
        int r = idx >> 2, seg = idx & 3;
        int rc = min(r, rows - 1);          // clamp: garbage rows feed masked outputs only
        long srow = GATHER ? (long)row_token[row0 + rc] : (long)(row0 + rc);
        aSrc[it] = A + srow * lda + seg * 8;
        aOff[it] = r * LDT + seg * 8;
        bSrc[it] = Bg + (long)(nb0 + r) * K + seg * 8;
        bOff[it] = r * LDT + seg * 8;
    }

    auto issue_tile = [&](int buf, int k0) {
        #pragma unroll
        for (int it = 0; it < 2; ++it) {
            async_ld_b128(aSrc[it] + k0, &As[buf][aOff[it]]);
            async_ld_b128(bSrc[it] + k0, &Bs[buf][bOff[it]]);
        }
    };

    const int wid = tid >> 5, lane = tid & 31;
    const int wm = wid >> 2, wn = wid & 3;    // 2x4 wave grid
    const int lhalf = lane >> 4, lmod = lane & 15;

    v8f acc[4][2];
    #pragma unroll
    for (int mi = 0; mi < 4; ++mi)
        #pragma unroll
        for (int ni = 0; ni < 2; ++ni) acc[mi][ni] = (v8f)(0.f);

    const int nk = K / BK;
    issue_tile(0, 0);
    for (int t = 0; t < nk; ++t) {
        const int buf = t & 1;
        wait_async();          // this wave's async copies into `buf` have landed
        __syncthreads();       // all waves' copies visible; previous reads of buf^1 done
        if (t + 1 < nk) issue_tile(buf ^ 1, (t + 1) * BK);   // overlap with WMMA below

        FragU a[4], b[2];
        #pragma unroll
        for (int mi = 0; mi < 4; ++mi) {
            int m = wm * 64 + mi * 16 + lmod;
            a[mi].q[0] = *(const uint4*)(&As[buf][m * LDT + lhalf * 8]);
            a[mi].q[1] = *(const uint4*)(&As[buf][m * LDT + 16 + lhalf * 8]);
        }
        #pragma unroll
        for (int ni = 0; ni < 2; ++ni) {
            int n = wn * 32 + ni * 16 + lmod;
            b[ni].q[0] = *(const uint4*)(&Bs[buf][n * LDT + lhalf * 8]);
            b[ni].q[1] = *(const uint4*)(&Bs[buf][n * LDT + 16 + lhalf * 8]);
        }
        #pragma unroll
        for (int mi = 0; mi < 4; ++mi)
            #pragma unroll
            for (int ni = 0; ni < 2; ++ni)
                acc[mi][ni] = __builtin_amdgcn_wmma_f32_16x16x32_bf16(
                    false, a[mi].v, false, b[ni].v, (short)0, acc[mi][ni], false, false);
        __syncthreads();       // done reading `buf` before next iter overwrites it
    }

    // --- epilogue ---
    #pragma unroll
    for (int ni = 0; ni < 2; ++ni) {
        const int colN = nb0 + wn * 32 + ni * 16 + lmod;
        const float bc = biasg[colN];
        #pragma unroll
        for (int mi = 0; mi < 4; ++mi) {
            #pragma unroll
            for (int r = 0; r < 8; ++r) {
                int lm = wm * 64 + mi * 16 + lhalf * 8 + r;
                if (lm >= rows) continue;
                long row = (long)row0 + lm;
                float v = acc[mi][ni][r] + bc;
                if (EPI == 0) {
                    outB[row * Nw + colN] = f2bf(tanhf(v));
                } else if (EPI == 1) {
                    atomicAdd(&outF[(long)rtok[lm] * Nw + colN], rgat[lm] * v);
                } else if (EPI == 2) {
                    float tv = tanhf(v + resid[row * Nw + colN]);
                    outF[row * Nw + colN] = tv;
                    outB[row * Nw + colN] = f2bf(tv);
                } else {
                    outF[row * Nw + colN] = v;
                }
            }
        }
    }
}

// ---------------- host orchestration ----------------
extern "C" void kernel_launch(void* const* d_in, const int* in_sizes, int n_in,
                              void* d_out, int out_size, void* d_ws, size_t ws_size,
                              hipStream_t stream) {
    const float* x    = (const float*)d_in[0];
    const float* wg   = (const float*)d_in[1];
    const float* W1   = (const float*)d_in[2];
    const float* b1   = (const float*)d_in[3];
    const float* W2   = (const float*)d_in[4];
    const float* b2   = (const float*)d_in[5];
    const float* rW1  = (const float*)d_in[6];
    const float* rb1  = (const float*)d_in[7];
    const float* rW2  = (const float*)d_in[8];
    const float* rb2  = (const float*)d_in[9];
    const float* oW   = (const float*)d_in[10];
    const float* ob   = (const float*)d_in[11];
    float* out = (float*)d_out;

    // --- workspace carve (256B aligned) ---
    size_t off = 0;
    auto alloc = [&](size_t sz) -> char* {
        size_t o = (off + 255) & ~(size_t)255;
        off = o + sz;
        return (char*)d_ws + o;
    };
    unsigned short* xb16 = (unsigned short*)alloc((size_t)N_TOK * C_IN * 2);
    unsigned short* w1t  = (unsigned short*)alloc((size_t)NE * C_IN * HIDN * 2);
    unsigned short* w2t  = (unsigned short*)alloc((size_t)NE * HIDN * HIDN * 2);
    unsigned short* rw1t = (unsigned short*)alloc((size_t)NRES * HIDN * HIDN * 2);
    unsigned short* rw2t = (unsigned short*)alloc((size_t)NRES * HIDN * HIDN * 2);
    unsigned short* owt  = (unsigned short*)alloc((size_t)HIDN * C_OUT * 2);
    unsigned short* h1   = (unsigned short*)alloc((size_t)N_TOK * KTOP * HIDN * 2);
    float* y_f32         = (float*)alloc((size_t)N_TOK * HIDN * 4);
    unsigned short* y_b16 = (unsigned short*)alloc((size_t)N_TOK * HIDN * 2);
    unsigned short* t_b16 = (unsigned short*)alloc((size_t)N_TOK * HIDN * 2);
    int*   top_idx  = (int*)alloc((size_t)N_TOK * KTOP * 4);
    float* top_gate = (float*)alloc((size_t)N_TOK * KTOP * 4);
    int*   row_token = (int*)alloc((size_t)N_TOK * KTOP * 4);
    float* row_gate  = (float*)alloc((size_t)N_TOK * KTOP * 4);
    int*   eoff = (int*)alloc(NE * 4);
    int*   ecnt = (int*)alloc(NE * 4);

    auto blocks = [](long total) { return (unsigned)((total + 255) / 256); };

    // --- precision conversion + weight transposition (bf16 tensor path) ---
    conv_f2b<<<blocks((long)N_TOK * C_IN), 256, 0, stream>>>(x, xb16, (long)N_TOK * C_IN);
    convT<<<blocks((long)NE * C_IN * HIDN), 256, 0, stream>>>(W1, w1t, C_IN, HIDN, (long)NE * C_IN * HIDN);
    convT<<<blocks((long)NE * HIDN * HIDN), 256, 0, stream>>>(W2, w2t, HIDN, HIDN, (long)NE * HIDN * HIDN);
    convT<<<blocks((long)NRES * HIDN * HIDN), 256, 0, stream>>>(rW1, rw1t, HIDN, HIDN, (long)NRES * HIDN * HIDN);
    convT<<<blocks((long)NRES * HIDN * HIDN), 256, 0, stream>>>(rW2, rw2t, HIDN, HIDN, (long)NRES * HIDN * HIDN);
    convT<<<blocks((long)HIDN * C_OUT), 256, 0, stream>>>(oW, owt, HIDN, C_OUT, (long)HIDN * C_OUT);

    // --- gating / stats / bucketing ---
    gating_kernel<<<N_TOK / 64, 256, 0, stream>>>(x, wg, top_idx, top_gate);
    moe_stats<<<1, 256, 0, stream>>>(top_idx, top_gate, out + (long)N_TOK * C_OUT);
    bucket_sort<<<1, 256, 0, stream>>>(top_idx, top_gate, row_token, row_gate, eoff, ecnt);
    zero_f32<<<blocks((long)N_TOK * HIDN), 256, 0, stream>>>(y_f32, (long)N_TOK * HIDN);

    // --- expert layer 1: gathered x @ W1[e], tanh -> h1 arena (bf16) ---
    gemm_wmma<true, 0><<<dim3(HIDN / BN, NE * 128), 256, 0, stream>>>(
        xb16, C_IN, w1t, (long)C_IN * HIDN, b1, HIDN, C_IN, HIDN,
        eoff, ecnt, 128, 0, row_token, nullptr, nullptr, h1, nullptr);

    // --- expert layer 2: h1 @ W2[e], gate-weighted scatter-add into y ---
    gemm_wmma<false, 1><<<dim3(HIDN / BN, NE * 128), 256, 0, stream>>>(
        h1, HIDN, w2t, (long)HIDN * HIDN, b2, HIDN, HIDN, HIDN,
        eoff, ecnt, 128, 0, row_token, row_gate, y_f32, nullptr, nullptr);

    conv_f2b<<<blocks((long)N_TOK * HIDN), 256, 0, stream>>>(y_f32, y_b16, (long)N_TOK * HIDN);

    // --- residual stack ---
    for (int i = 0; i < NRES; ++i) {
        gemm_wmma<false, 0><<<dim3(HIDN / BN, N_TOK / BM), 256, 0, stream>>>(
            y_b16, HIDN, rw1t + (long)i * HIDN * HIDN, 0, rb1 + i * HIDN, 0, HIDN, HIDN,
            nullptr, nullptr, N_TOK / BM, N_TOK, nullptr, nullptr, nullptr, t_b16, nullptr);
        gemm_wmma<false, 2><<<dim3(HIDN / BN, N_TOK / BM), 256, 0, stream>>>(
            t_b16, HIDN, rw2t + (long)i * HIDN * HIDN, 0, rb2 + i * HIDN, 0, HIDN, HIDN,
            nullptr, nullptr, N_TOK / BM, N_TOK, nullptr, nullptr, y_f32, y_b16, y_f32);
    }

    // --- final projection ---
    gemm_wmma<false, 3><<<dim3(C_OUT / BN, N_TOK / BM), 256, 0, stream>>>(
        y_b16, HIDN, owt, 0, ob, 0, HIDN, C_OUT,
        nullptr, nullptr, N_TOK / BM, N_TOK, nullptr, nullptr, out, nullptr, nullptr);

    (void)in_sizes; (void)n_in; (void)out_size; (void)ws_size;
}
